// MaxCosineSimilarityBlock_72679436583773
// MI455X (gfx1250) — matrix operand order repaired and checked by reading, not compiled, not run on hardware
//
#include <hip/hip_runtime.h>
#include <hip/hip_bf16.h>

typedef _Float16 v16h __attribute__((ext_vector_type(16)));
typedef _Float16 h8   __attribute__((ext_vector_type(8)));
typedef float    v8f  __attribute__((ext_vector_type(8)));

#define TDIM 1024
#define NDIM 512
#define SDIM 64
#define CDIM 8
#define TTILE 128

__device__ __forceinline__ float relu1(float v) {
    // single v_med3_num_f32: median(v, 0, +inf) == clamp to [0, inf)
    return __builtin_amdgcn_fmed3f(v, 0.0f, __builtin_inff());
}

// ---------------------------------------------------------------------------
// Kernel 1: normalize shapelets -> f16, stored in WMMA B-fragment layout:
//   frag[c][j][kc][lane][16 halves],  c<8, j<32 (n-tile), kc<2 (k chunk), lane<32
//   lane l (l<16):  column N = j*16+l, halves = K[kc*32 + 0..15]
//   lane l (l>=16): column N = j*16+(l-16), halves = K[kc*32 + 16..31]
// ---------------------------------------------------------------------------
__global__ __launch_bounds__(256)
void shapelet_prep_kernel(const float* __restrict__ sh, _Float16* __restrict__ frag) {
    int tid = blockIdx.x * 256 + threadIdx.x;       // 0..4095  => (c,n)
    if (tid >= CDIM * NDIM) return;
    int c = tid >> 9;
    int n = tid & (NDIM - 1);
    const float* row = sh + (size_t)tid * SDIM;

    float ss = 0.f;
    #pragma unroll
    for (int s = 0; s < SDIM; ++s) ss += row[s] * row[s];
    float inv = 1.f / fmaxf(sqrtf(ss), 1e-8f);

    int j = n >> 4, l = n & 15;
    #pragma unroll
    for (int kc = 0; kc < 2; ++kc) {
        #pragma unroll
        for (int h = 0; h < 2; ++h) {
            size_t base = ((((size_t)c * 32 + j) * 2 + kc) * 32 + (h * 16 + l)) * 16;
            #pragma unroll
            for (int p = 0; p < 16; ++p)
                frag[base + p] = (_Float16)(row[kc * 32 + h * 16 + p] * inv);
        }
    }
}

// ---------------------------------------------------------------------------
// Kernel 2: per block: one (b,c), 128 consecutive t. 8 waves x 16 t each.
//   LDS: raw x slice -> sliding-window inv-norms -> normalized f16 A rows.
//   Each wave: A fragments loaded once, loop 32 n-tiles of B, 2 WMMAs (K=64),
//   ReLU (v_med3), nontemporal coalesced stores (write-once 537MB stream).
// ---------------------------------------------------------------------------
__global__ __launch_bounds__(256)
void cosine_wmma_kernel(const float* __restrict__ x,
                        const _Float16* __restrict__ frag,
                        float* __restrict__ out) {
    __shared__ __align__(16) _Float16 Arows[TTILE * SDIM];   // 16 KB
    __shared__ float raw[TTILE + SDIM];                      // 192 floats
    __shared__ float invn[TTILE];

    const int tid = threadIdx.x;
    const int blk = blockIdx.x;
    const int tc  = blk & 7;            // which 128-wide t chunk
    const int bc  = blk >> 3;           // b*8 + c
    const int c   = bc & 7;
    const int t0  = tc * TTILE;

    const float* xrow = x + (size_t)bc * TDIM;

    // window t covers x[t-31 .. t+32]; raw[i] = x[t0 - 31 + i], zero-padded
    for (int i = tid; i < TTILE + SDIM; i += 256) {
        int g = t0 - 31 + i;
        raw[i] = (g >= 0 && g < TDIM) ? xrow[g] : 0.f;
    }
    __syncthreads();

    // per-window inverse L2 norm (window t0+j = raw[j .. j+63])
    for (int j = tid; j < TTILE; j += 256) {
        float ss = 0.f;
        #pragma unroll 8
        for (int s = 0; s < SDIM; ++s) { float v = raw[j + s]; ss += v * v; }
        invn[j] = 1.f / fmaxf(sqrtf(ss), 1e-8f);
    }
    __syncthreads();

    // normalized f16 window rows, row-major, 64 halves per row
    for (int idx = tid; idx < TTILE * SDIM; idx += 256) {
        int r = idx >> 6, s = idx & 63;
        Arows[idx] = (_Float16)(raw[r + s] * invn[r]);
    }
    __syncthreads();

    const int wave = tid >> 5;
    const int lane = tid & 31;
    const int tw   = wave * 16;
    const int rowA = tw + (lane & 15);
    const int hh   = lane >> 4;

    // A fragments (16x32 f16 layout): lane<16 gets K {kc*32+0..15},
    // lane>=16 gets K {kc*32+16..31}
    v16h a0, a1;
    {
        const _Float16* base = Arows + rowA * SDIM;
        h8 lo0 = *(const h8*)(base +  0 + hh * 16);
        h8 hi0 = *(const h8*)(base +  8 + hh * 16);
        h8 lo1 = *(const h8*)(base + 32 + hh * 16);
        h8 hi1 = *(const h8*)(base + 40 + hh * 16);
        a0 = __builtin_shufflevector(lo0, hi0, 0,1,2,3,4,5,6,7,8,9,10,11,12,13,14,15);
        a1 = __builtin_shufflevector(lo1, hi1, 0,1,2,3,4,5,6,7,8,9,10,11,12,13,14,15);
    }

    const _Float16* fb = frag + (size_t)c * (NDIM * SDIM);
    float* obase = out + ((size_t)bc * TDIM + (t0 + tw)) * NDIM + (lane & 15);
    const int rshift = hh * 8;          // high lane half owns rows M+8

    for (int j = 0; j < NDIM / 16; ++j) {
        v16h b0 = *(const v16h*)(fb + ((size_t)(j * 2 + 0) * 32 + lane) * 16);
        v16h b1 = *(const v16h*)(fb + ((size_t)(j * 2 + 1) * 32 + lane) * 16);
        v8f acc = {};
        acc = __builtin_amdgcn_wmma_f32_16x16x32_f16(false, a0, false, b0,
                                                     (short)0, acc, false, false);
        acc = __builtin_amdgcn_wmma_f32_16x16x32_f16(false, a1, false, b1,
                                                     (short)0, acc, false, false);
        float* o = obase + j * 16;
        #pragma unroll
        for (int r = 0; r < 8; ++r) {
            __builtin_nontemporal_store(relu1(acc[r]), o + (size_t)(r + rshift) * NDIM);
        }
    }
}

// ---------------------------------------------------------------------------
extern "C" void kernel_launch(void* const* d_in, const int* in_sizes, int n_in,
                              void* d_out, int out_size, void* d_ws, size_t ws_size,
                              hipStream_t stream) {
    const float* x  = (const float*)d_in[0];       // (32, 8, 1024) f32
    const float* sh = (const float*)d_in[1];       // (8, 512, 64)  f32
    float* out = (float*)d_out;                    // (32, 8, 1024, 512) f32
    _Float16* frag = (_Float16*)d_ws;              // needs 8*512*64*2 = 512 KB

    shapelet_prep_kernel<<<(CDIM * NDIM + 255) / 256, 256, 0, stream>>>(sh, frag);

    // 32*8 (b,c) pairs * 8 t-chunks of 128 = 2048 blocks
    cosine_wmma_kernel<<<2048, 256, 0, stream>>>(x, frag, out);
}
